// PerformerWithFFNBlock_23733989277969
// MI455X (gfx1250) — compile-verified, matrix operand order checked by reading, not run on hardware
//
#include <hip/hip_runtime.h>
#include <cstdint>

// ---------------------------------------------------------------------------
// Types
// ---------------------------------------------------------------------------
typedef __bf16 bf16;
typedef __attribute__((ext_vector_type(16))) __bf16 v16bf;
typedef __attribute__((ext_vector_type(8)))  float  v8f;
typedef __attribute__((ext_vector_type(4)))  int    v4i;

// ---------------------------------------------------------------------------
// Problem dimensions (compile-time, match the reference)
// ---------------------------------------------------------------------------
constexpr int  B_  = 4;
constexpr int  N_  = 4096;
constexpr int  D_  = 1024;
constexpr int  H_  = 16;
constexpr int  DH_ = 64;
constexpr int  M_  = 256;
constexpr int  F_  = 2048;
constexpr long T_  = (long)B_ * N_;               // 16384 tokens

constexpr float DN_    = 0.35355339059327373f;    // 64^-0.25
constexpr float RATIO_ = 0.0625f;                 // 256^-0.5
constexpr float KEPS_  = 1e-3f;
constexpr float LNEPS_ = 1e-5f;

// ---------------------------------------------------------------------------
// CDNA5 async global->LDS (signature probe-confirmed in round 1/2:
// (v4i AS1* gsrc, v4i AS3* ldsdst, imm offset, imm cpol); emits
// global_load_async_to_lds_b128 + s_wait_asynccnt)
// ---------------------------------------------------------------------------
#ifndef __has_builtin
#define __has_builtin(x) 0
#endif
#if __has_builtin(__builtin_amdgcn_global_load_async_to_lds_b128)
#define USE_ASYNC_LDS 1
#else
#define USE_ASYNC_LDS 0
#endif

// ---------------------------------------------------------------------------
// WMMA GEMM:  C[r,c] = sum_k A[r,k] * W[c,k]   (i.e. C = A * W^T)
// A: row-major [rows, K] (lda), W: row-major [cols, K] (ldw), C fp32.
// Batched via blockIdx.z = zo*zInner + zi with separate strides per operand.
// Tile: BM=128 x BN=64, BK=64, double-buffered LDS; 256 threads = 8 wave32s,
// each wave owns a 32x32 output tile (2x2 WMMA accumulators, 8 WMMA/K-tile).
// Async DMA for K-tile t+1 overlaps WMMA on K-tile t via s_wait_asynccnt 6.
// ---------------------------------------------------------------------------
#define BM 128
#define BN 64
#define BK 64
#define LDSTR 72        // LDS row stride in bf16 (144B = 9*16B: aligned, padded)
#define STAGE_CHUNKS 6  // async b128 loads issued per thread per stage (4 A + 2 B)

__device__ __forceinline__ void copy_chunk(const bf16* __restrict__ src, bf16* dst) {
#if USE_ASYNC_LDS
  __builtin_amdgcn_global_load_async_to_lds_b128(
      (__attribute__((address_space(1))) v4i*)(uintptr_t)src,
      (__attribute__((address_space(3))) v4i*)(uintptr_t)dst, 0, 0);
#else
  *(uint4*)dst = *(const uint4*)src;
#endif
}

// Stage one BMxBK A-tile and one BNxBK B-tile into LDS (8 chunks of 8 bf16/row)
__device__ __forceinline__ void stage(const bf16* __restrict__ A, int lda,
                                      const bf16* __restrict__ W, int ldw,
                                      bf16* la, bf16* lb, int tid) {
  for (int c = tid; c < BM * (BK / 8); c += 256) {   // 1024 chunks -> 4/thread
    const int r = c >> 3, q = (c & 7) << 3;
    copy_chunk(A + (long)r * lda + q, la + r * LDSTR + q);
  }
  for (int c = tid; c < BN * (BK / 8); c += 256) {   // 512 chunks -> 2/thread
    const int r = c >> 3, q = (c & 7) << 3;
    copy_chunk(W + (long)r * ldw + q, lb + r * LDSTR + q);
  }
}

#if USE_ASYNC_LDS
__device__ __forceinline__ void wait_async_pipe() {
#if __has_builtin(__builtin_amdgcn_s_wait_asynccnt)
  __builtin_amdgcn_s_wait_asynccnt(STAGE_CHUNKS);
#else
  asm volatile("s_wait_asynccnt 6" ::: "memory");
#endif
}
__device__ __forceinline__ void wait_async_all() {
#if __has_builtin(__builtin_amdgcn_s_wait_asynccnt)
  __builtin_amdgcn_s_wait_asynccnt(0);
#else
  asm volatile("s_wait_asynccnt 0" ::: "memory");
#endif
}
#endif

__device__ __forceinline__ v16bf frag_load(const bf16* lds, int row0, int kk, int lane) {
  // 16-bit A-operand layout: lane (0..15) -> matrix row, lane>>4 selects the
  // K half: VGPR0-3 <- K[kh*8 .. +7], VGPR4-7 <- K[16+kh*8 .. +7].
  const int r  = row0 + (lane & 15);
  const int kh = (lane >> 4) << 3;
  const bf16* p = lds + r * LDSTR + kk * 32 + kh;
  union { uint4 u[2]; v16bf v; } f;
  f.u[0] = *(const uint4*)(p);
  f.u[1] = *(const uint4*)(p + 16);
  return f.v;
}

template <bool TRANS_C, bool HAS_BIAS>
__global__ __launch_bounds__(256) void gemm_kernel(
    const bf16* __restrict__ A, long sAo, long sAi, int lda,
    const bf16* __restrict__ W, long sWo, long sWi, int ldw,
    float* __restrict__ C, long sCo, long sCi, int ldc,
    const float* __restrict__ bias, int K, int zInner) {
  __shared__ bf16 lA[2][BM * LDSTR];
  __shared__ bf16 lB[2][BN * LDSTR];

  const int z  = blockIdx.z;
  const int zo = z / zInner;
  const int zi = z % zInner;
  A += zo * sAo + zi * sAi;
  W += zo * sWo + zi * sWi;
  C += zo * sCo + zi * sCi;

  const long row0 = (long)blockIdx.y * BM;
  const long col0 = (long)blockIdx.x * BN;
  A += row0 * lda;
  W += col0 * ldw;

  const int tid  = threadIdx.x;
  const int lane = tid & 31;
  const int wave = tid >> 5;
  const int wm = (wave >> 1) << 5;   // wave row offset within block tile
  const int wn = (wave & 1) << 5;    // wave col offset within block tile

  v8f acc[2][2] = {};

  const int ksteps = K / BK;
  stage(A, lda, W, ldw, lA[0], lB[0], tid);   // prologue: tile 0

  int buf = 0;
  for (int t = 0; t < ksteps; ++t, buf ^= 1) {
    const bool has_next = (t + 1 < ksteps);
    if (has_next)
      stage(A + (t + 1) * BK, lda, W + (t + 1) * BK, ldw,
            lA[buf ^ 1], lB[buf ^ 1], tid);
#if USE_ASYNC_LDS
    // Loads retire in order: allowing STAGE_CHUNKS outstanding guarantees the
    // current buffer has landed while tile t+1 streams in under the WMMAs.
    if (has_next) wait_async_pipe();
    else          wait_async_all();
#endif
    __syncthreads();

    const bf16* la = lA[buf];
    const bf16* lb = lB[buf];
#pragma unroll
    for (int kk = 0; kk < BK / 32; ++kk) {
      const v16bf a0 = frag_load(la, wm, kk, lane);
      const v16bf a1 = frag_load(la, wm + 16, kk, lane);
      const v16bf b0 = frag_load(lb, wn, kk, lane);
      const v16bf b1 = frag_load(lb, wn + 16, kk, lane);
      acc[0][0] = __builtin_amdgcn_wmma_f32_16x16x32_bf16(false, a0, false, b0,
                                                          (short)0, acc[0][0], false, false);
      acc[0][1] = __builtin_amdgcn_wmma_f32_16x16x32_bf16(false, a0, false, b1,
                                                          (short)0, acc[0][1], false, false);
      acc[1][0] = __builtin_amdgcn_wmma_f32_16x16x32_bf16(false, a1, false, b0,
                                                          (short)0, acc[1][0], false, false);
      acc[1][1] = __builtin_amdgcn_wmma_f32_16x16x32_bf16(false, a1, false, b1,
                                                          (short)0, acc[1][1], false, false);
    }
    __syncthreads();
  }

  // C/D layout: VGPR v, lanes 0-15 -> M=v, N=lane; lanes 16-31 -> M=v+8.
  const int lr = (lane >> 4) << 3;
  const int lc = lane & 15;
#pragma unroll
  for (int i = 0; i < 2; ++i)
#pragma unroll
    for (int j = 0; j < 2; ++j)
#pragma unroll
      for (int v = 0; v < 8; ++v) {
        const long gr = row0 + wm + i * 16 + lr + v;
        const long gc = col0 + wn + j * 16 + lc;
        float val = acc[i][j][v];
        if (HAS_BIAS) val += bias[gc];
        if (TRANS_C) C[gc * (long)ldc + gr] = val;
        else         C[gr * (long)ldc + gc] = val;
      }
}

// ---------------------------------------------------------------------------
// Elementwise / reduction kernels
// ---------------------------------------------------------------------------
__global__ void k_convert(const float* __restrict__ in, bf16* __restrict__ out, long n) {
  for (long i = blockIdx.x * (long)blockDim.x + threadIdx.x; i < n;
       i += (long)gridDim.x * blockDim.x)
    out[i] = (bf16)in[i];
}

// [T_, D_] f32 -> [D_, T_] bf16 (used for V -> vT)
__global__ void k_transpose_convert(const float* __restrict__ in, bf16* __restrict__ out) {
  const long n = T_ * (long)D_;
  for (long i = blockIdx.x * (long)blockDim.x + threadIdx.x; i < n;
       i += (long)gridDim.x * blockDim.x) {
    const long r = i / D_;
    const int  c = (int)(i % D_);
    out[(long)c * T_ + r] = (bf16)in[i];
  }
}

// diag[h*T + r] = 0.5*dn^2 * sum_d X[r, h*64+d]^2
__global__ void k_diag(const float* __restrict__ X, float* __restrict__ dg) {
  const long n = (long)H_ * T_;
  for (long i = blockIdx.x * (long)blockDim.x + threadIdx.x; i < n;
       i += (long)gridDim.x * blockDim.x) {
    const int  h = (int)(i / T_);
    const long r = i % T_;
    const float* p = X + r * D_ + h * DH_;
    float s = 0.f;
    for (int d = 0; d < DH_; ++d) s += p[d] * p[d];
    dg[i] = 0.5f * DN_ * DN_ * s;
  }
}

__device__ __forceinline__ unsigned f_enc(float f) {
  unsigned u = __float_as_uint(f);
  return (u & 0x80000000u) ? ~u : (u | 0x80000000u);
}
__device__ __forceinline__ float f_dec(unsigned k) {
  unsigned u = (k & 0x80000000u) ? (k ^ 0x80000000u) : ~k;
  return __uint_as_float(u);
}

__global__ void k_set_u32(unsigned* p, unsigned v) { *p = v; }

__global__ __launch_bounds__(256) void k_max(const float* __restrict__ dd, long n,
                                             unsigned* __restrict__ key) {
  __shared__ float red[256];
  const int tid = threadIdx.x;
  float m = -3.4e38f;
  for (long i = blockIdx.x * (long)blockDim.x + tid; i < n;
       i += (long)gridDim.x * blockDim.x)
    m = fmaxf(m, DN_ * dd[i]);
  red[tid] = m;
  __syncthreads();
  for (int o = 128; o > 0; o >>= 1) {
    if (tid < o) red[tid] = fmaxf(red[tid], red[tid + o]);
    __syncthreads();
  }
  if (tid == 0) atomicMax(key, f_enc(red[0]));
}

// qp[idx, m] = ratio*(exp(dn*dd - diag - rowmax(dn*dd)) + eps), idx over H_*T_
__global__ void k_qp(const float* __restrict__ dd, const float* __restrict__ dg,
                     bf16* __restrict__ qp) {
  const long n = (long)H_ * T_;
  const long idx = blockIdx.x * (long)blockDim.x + threadIdx.x;
  if (idx >= n) return;
  const float* row = dd + idx * M_;
  float mx = -3.4e38f;
  for (int m = 0; m < M_; ++m) mx = fmaxf(mx, DN_ * row[m]);
  const float d = dg[idx];
  bf16* o = qp + idx * M_;
  for (int m = 0; m < M_; ++m)
    o[m] = (bf16)(RATIO_ * (expf(DN_ * row[m] - d - mx) + KEPS_));
}

// kpT[(h*M+m)*T + r] with global stab
__global__ void k_kp(const float* __restrict__ dd, const float* __restrict__ dg,
                     const unsigned* __restrict__ key, bf16* __restrict__ kpT) {
  const long n = (long)H_ * T_;
  const long idx = blockIdx.x * (long)blockDim.x + threadIdx.x;
  if (idx >= n) return;
  const float stab = f_dec(*key);
  const int  h = (int)(idx / T_);
  const long r = idx % T_;
  const float* row = dd + idx * M_;
  const float d = dg[idx];
  for (int m = 0; m < M_; ++m)
    kpT[((long)(h * M_ + m)) * T_ + r] =
        (bf16)(RATIO_ * (expf(DN_ * row[m] - d - stab) + KEPS_));
}

// ksum[(b*H+h)*M+m] = sum_n kpT[(h*M+m)*T + b*N + n]
__global__ void k_ksum(const bf16* __restrict__ kpT, float* __restrict__ ks) {
  const long n = (long)B_ * H_ * M_;
  const long idx = blockIdx.x * (long)blockDim.x + threadIdx.x;
  if (idx >= n) return;
  const int b = (int)(idx / (H_ * M_));
  const int h = (int)((idx / M_) % H_);
  const int m = (int)(idx % M_);
  const bf16* p = kpT + ((long)(h * M_ + m)) * T_ + (long)b * N_;
  float s = 0.f;
  for (int t = 0; t < N_; ++t) s += (float)p[t];
  ks[idx] = s;
}

// dinv[h*T + r] = 1 / sum_m qp[idx,m] * ksum[(b*H+h)*M+m]
__global__ void k_dinv(const bf16* __restrict__ qp, const float* __restrict__ ks,
                       float* __restrict__ dinv) {
  const long n = (long)H_ * T_;
  const long idx = blockIdx.x * (long)blockDim.x + threadIdx.x;
  if (idx >= n) return;
  const int  h = (int)(idx / T_);
  const long r = idx % T_;
  const int  b = (int)(r / N_);
  const bf16*  q = qp + idx * M_;
  const float* s = ks + ((long)b * H_ + h) * M_;
  float acc = 0.f;
  for (int m = 0; m < M_; ++m) acc += (float)q[m] * s[m];
  dinv[idx] = 1.0f / acc;
}

__global__ void k_attn_scale(const float* __restrict__ a32, const float* __restrict__ dinv,
                             bf16* __restrict__ a16) {
  const long n = T_ * (long)D_;
  for (long i = blockIdx.x * (long)blockDim.x + threadIdx.x; i < n;
       i += (long)gridDim.x * blockDim.x) {
    const long r = i / D_;
    const int  c = (int)(i % D_);
    const int  h = c >> 6;
    a16[i] = (bf16)(a32[i] * dinv[(long)h * T_ + r]);
  }
}

__global__ void k_gelu(const float* __restrict__ in, bf16* __restrict__ out, long n) {
  for (long i = blockIdx.x * (long)blockDim.x + threadIdx.x; i < n;
       i += (long)gridDim.x * blockDim.x) {
    const float x = in[i];
    out[i] = (bf16)(0.5f * x * (1.0f + erff(x * 0.7071067811865476f)));
  }
}

// LayerNorm over rows of (A + Bv); writes fp32 and optionally bf16.
__global__ __launch_bounds__(256) void k_ln(const float* __restrict__ A,
                                            const float* __restrict__ Bv,
                                            const float* __restrict__ g,
                                            const float* __restrict__ be,
                                            float* __restrict__ o32,
                                            bf16* __restrict__ o16) {
  __shared__ float red[256];
  const long row = blockIdx.x;
  const int  tid = threadIdx.x;
  const float* a = A  + row * D_;
  const float* b = Bv + row * D_;

  float s = 0.f;
  for (int c = tid; c < D_; c += 256) s += a[c] + b[c];
  red[tid] = s;
  __syncthreads();
  for (int o = 128; o > 0; o >>= 1) { if (tid < o) red[tid] += red[tid + o]; __syncthreads(); }
  const float mu = red[0] * (1.0f / D_);
  __syncthreads();

  float v = 0.f;
  for (int c = tid; c < D_; c += 256) { const float t = a[c] + b[c] - mu; v += t * t; }
  red[tid] = v;
  __syncthreads();
  for (int o = 128; o > 0; o >>= 1) { if (tid < o) red[tid] += red[tid + o]; __syncthreads(); }
  const float inv = rsqrtf(red[0] * (1.0f / D_) + LNEPS_);

  for (int c = tid; c < D_; c += 256) {
    const float y = (a[c] + b[c] - mu) * inv * g[c] + be[c];
    o32[row * D_ + c] = y;
    if (o16) o16[row * D_ + c] = (bf16)y;
  }
}

// ---------------------------------------------------------------------------
// Host launcher
// ---------------------------------------------------------------------------
extern "C" void kernel_launch(void* const* d_in, const int* in_sizes, int n_in,
                              void* d_out, int out_size, void* d_ws, size_t ws_size,
                              hipStream_t stream) {
  (void)in_sizes; (void)n_in; (void)out_size; (void)ws_size;

  const float* src   = (const float*)d_in[0];
  const float* Wq    = (const float*)d_in[1];
  const float* Wk    = (const float*)d_in[2];
  const float* Wv    = (const float*)d_in[3];
  const float* Wout  = (const float*)d_in[4];
  const float* b_out = (const float*)d_in[5];
  const float* proj  = (const float*)d_in[6];
  const float* g1    = (const float*)d_in[7];
  const float* beta1 = (const float*)d_in[8];
  const float* W1    = (const float*)d_in[9];
  const float* b1    = (const float*)d_in[10];
  const float* W2    = (const float*)d_in[11];
  const float* b2    = (const float*)d_in[12];
  const float* g2    = (const float*)d_in[13];
  const float* beta2 = (const float*)d_in[14];
  float* out = (float*)d_out;

  // ---- workspace carve-up (lifetime-aliased slots, ~437 MB total) ----
  char* wsb = (char*)d_ws;
  size_t off = 0;
  auto take = [&](size_t bytes) -> char* {
    char* p = wsb + off;
    off += (bytes + 255) & ~(size_t)255;
    return p;
  };
  const size_t MB = 1024ull * 1024ull;

  bf16* X16    = (bf16*)take((size_t)T_ * D_ * 2);
  bf16* Wq16   = (bf16*)take((size_t)D_ * D_ * 2);
  bf16* Wk16   = (bf16*)take((size_t)D_ * D_ * 2);
  bf16* Wv16   = (bf16*)take((size_t)D_ * D_ * 2);
  bf16* Wout16 = (bf16*)take((size_t)D_ * D_ * 2);
  bf16* W116   = (bf16*)take((size_t)F_ * D_ * 2);
  bf16* W216   = (bf16*)take((size_t)D_ * F_ * 2);
  bf16* proj16 = (bf16*)take((size_t)M_ * DH_ * 2);
  float* diag_q = (float*)take((size_t)H_ * T_ * 4);
  float* diag_k = (float*)take((size_t)H_ * T_ * 4);
  float* ksum   = (float*)take((size_t)B_ * H_ * M_ * 4);
  float* dinv   = (float*)take((size_t)H_ * T_ * 4);
  unsigned* kstab = (unsigned*)take(256);
  char* S0  = take(64 * MB);
  char* S1  = take(64 * MB);
  char* S2  = take(64 * MB);
  char* S3  = take(64 * MB);
  char* S45 = take(128 * MB);

  float* Q32    = (float*)S0;                 // QKV fp32
  float* K32    = (float*)S1;
  float* V32    = (float*)S2;
  bf16*  Q16    = (bf16*)S3;                  // [0,32MB)
  bf16*  K16    = (bf16*)(S3 + 32 * MB);      // [32,64MB)
  bf16*  vT16   = (bf16*)S45;                 // [0,32MB) of S45
  float* dd_q   = (float*)S0;                 // overwrites Q32
  float* dd_k   = (float*)S1;                 // overwrites K32
  bf16*  qp16   = (bf16*)S2;                  // overwrites V32
  bf16*  kpT16  = (bf16*)(S2 + 32 * MB);
  float* ctxT32 = (float*)S0;                 // overwrites dd_q (4MB)
  bf16*  ctxT16 = (bf16*)(S0 + 8 * MB);
  float* attn32 = (float*)S1;                 // overwrites dd_k
  bf16*  attn16 = (bf16*)S3;                  // overwrites Q16/K16
  float* src2   = (float*)S0;                 // overwrites ctx*
  float* xbuf   = (float*)S1;                 // overwrites attn32
  bf16*  x16    = (bf16*)S2;                  // overwrites qp16
  float* ffh32  = (float*)S45;                // overwrites vT16 (full 128MB)
  bf16*  ffh16  = (bf16*)S0;                  // overwrites src2
  float* y32    = (float*)S3;                 // overwrites attn16

  const dim3 blk(256);

  // ---- 0) fp32 -> bf16 conversions ----
  k_convert<<<dim3(4096), blk, 0, stream>>>(src,  X16,    T_ * (long)D_);
  k_convert<<<dim3(1024), blk, 0, stream>>>(Wq,   Wq16,   (long)D_ * D_);
  k_convert<<<dim3(1024), blk, 0, stream>>>(Wk,   Wk16,   (long)D_ * D_);
  k_convert<<<dim3(1024), blk, 0, stream>>>(Wv,   Wv16,   (long)D_ * D_);
  k_convert<<<dim3(1024), blk, 0, stream>>>(Wout, Wout16, (long)D_ * D_);
  k_convert<<<dim3(2048), blk, 0, stream>>>(W1,   W116,   (long)F_ * D_);
  k_convert<<<dim3(2048), blk, 0, stream>>>(W2,   W216,   (long)D_ * F_);
  k_convert<<<dim3(64),   blk, 0, stream>>>(proj, proj16, (long)M_ * DH_);

  // ---- 1) Q/K/V projections: [T,D] = X16 * W^T ----
  const dim3 gP(D_ / BN, (int)(T_ / BM), 1);
  gemm_kernel<false, false><<<gP, blk, 0, stream>>>(X16, 0, 0, D_, Wq16, 0, 0, D_,
                                                    Q32, 0, 0, D_, nullptr, D_, 1);
  gemm_kernel<false, false><<<gP, blk, 0, stream>>>(X16, 0, 0, D_, Wk16, 0, 0, D_,
                                                    K32, 0, 0, D_, nullptr, D_, 1);
  gemm_kernel<false, false><<<gP, blk, 0, stream>>>(X16, 0, 0, D_, Wv16, 0, 0, D_,
                                                    V32, 0, 0, D_, nullptr, D_, 1);

  // ---- 2) bf16 copies + per-head diagonals ----
  k_convert<<<dim3(4096), blk, 0, stream>>>(Q32, Q16, T_ * (long)D_);
  k_convert<<<dim3(4096), blk, 0, stream>>>(K32, K16, T_ * (long)D_);
  k_transpose_convert<<<dim3(4096), blk, 0, stream>>>(V32, vT16);
  k_diag<<<dim3(1024), blk, 0, stream>>>(Q32, diag_q);
  k_diag<<<dim3(1024), blk, 0, stream>>>(K32, diag_k);

  // ---- 3) feature-map GEMMs: dd[h][t][m] = Q_head * proj^T (raw, dn later) ----
  const dim3 gF(M_ / BN, (int)(T_ / BM), H_);
  gemm_kernel<false, false><<<gF, blk, 0, stream>>>(
      Q16, 0, DH_, D_, proj16, 0, 0, DH_, dd_q, 0, T_ * (long)M_, M_, nullptr, DH_, H_);
  gemm_kernel<false, false><<<gF, blk, 0, stream>>>(
      K16, 0, DH_, D_, proj16, 0, 0, DH_, dd_k, 0, T_ * (long)M_, M_, nullptr, DH_, H_);

  // ---- 4) stabilizers + positive feature maps ----
  k_set_u32<<<dim3(1), dim3(1), 0, stream>>>(kstab, 0u);
  k_max<<<dim3(4096), blk, 0, stream>>>(dd_k, (long)H_ * T_ * M_, kstab);
  k_qp<<<dim3(1024), blk, 0, stream>>>(dd_q, diag_q, qp16);
  k_kp<<<dim3(1024), blk, 0, stream>>>(dd_k, diag_k, kstab, kpT16);
  k_ksum<<<dim3(64), blk, 0, stream>>>(kpT16, ksum);

  // ---- 5) context[b,h] = kp^T * v  (stored transposed -> ctxT[b,h,d,m]) ----
  const dim3 gC(DH_ / BN, M_ / BM, B_ * H_);
  gemm_kernel<true, false><<<gC, blk, 0, stream>>>(
      kpT16, N_, (long)M_ * T_, (int)T_,
      vT16,  N_, (long)DH_ * T_, (int)T_,
      ctxT32, (long)H_ * DH_ * M_, (long)DH_ * M_, M_, nullptr, N_, H_);
  k_convert<<<dim3(1024), blk, 0, stream>>>(ctxT32, ctxT16, (long)B_ * H_ * DH_ * M_);
  k_dinv<<<dim3(1024), blk, 0, stream>>>(qp16, ksum, dinv);

  // ---- 6) attn[b,h] = qp * ctx  -> attn32[T,D] (head-merged columns) ----
  const dim3 gA(1, N_ / BM, B_ * H_);
  gemm_kernel<false, false><<<gA, blk, 0, stream>>>(
      qp16, (long)N_ * M_, T_ * (long)M_, M_,
      ctxT16, (long)H_ * DH_ * M_, (long)DH_ * M_, M_,
      attn32, (long)N_ * D_, DH_, D_, nullptr, M_, H_);
  k_attn_scale<<<dim3(8192), blk, 0, stream>>>(attn32, dinv, attn16);

  // ---- 7) output projection + bias ----
  gemm_kernel<false, true><<<gP, blk, 0, stream>>>(attn16, 0, 0, D_, Wout16, 0, 0, D_,
                                                   src2, 0, 0, D_, b_out, D_, 1);

  // ---- 8) residual + LN1 -> x (fp32 + bf16) ----
  k_ln<<<dim3((int)T_), blk, 0, stream>>>(src, src2, g1, beta1, xbuf, x16);

  // ---- 9) FFN ----
  const dim3 gH(F_ / BN, (int)(T_ / BM), 1);
  gemm_kernel<false, true><<<gH, blk, 0, stream>>>(x16, 0, 0, D_, W116, 0, 0, D_,
                                                   ffh32, 0, 0, F_, b1, D_, 1);
  k_gelu<<<dim3(8192), blk, 0, stream>>>(ffh32, ffh16, T_ * (long)F_);
  gemm_kernel<false, true><<<gP, blk, 0, stream>>>(ffh16, 0, 0, F_, W216, 0, 0, F_,
                                                   y32, 0, 0, D_, b2, F_, 1);

  // ---- 10) residual + LN2 -> out ----
  k_ln<<<dim3((int)T_), blk, 0, stream>>>(xbuf, y32, g2, beta2, out, nullptr);
}
